// Encoder_71554155151954
// MI455X (gfx1250) — matrix operand order repaired and checked by reading, not compiled, not run on hardware
//
#include <hip/hip_runtime.h>
#include <math.h>

typedef __attribute__((ext_vector_type(16))) __bf16 v16bf;
typedef __attribute__((ext_vector_type(8)))  __bf16 v8bf;
typedef __attribute__((ext_vector_type(8)))  float  v8f;

// 8 waves (= 8 experts) x 32 lanes. BM = 8 experts x 16 ch = 128, BN = 128, BK = 32.
#define BR 8
#define BN 128
#define BK 32
#define LSTR 40      // LDS row stride in bf16 (80 B, 16-B aligned)
#define KD_MAX 1664  // >= max Kdim (1600)

// ---- one-time weight pre-convert: f32 [Mrows,Kdim] -> bf16 [Mrows,KdimPad], zero-padded
__global__ __launch_bounds__(256) void convert_weights(
    const float* __restrict__ w, __bf16* __restrict__ wbf,
    int Mrows, int Kdim, int KdimPad)
{
    int i = blockIdx.x * 256 + threadIdx.x;
    if (i >= Mrows * KdimPad) return;
    int row = i / KdimPad;
    int kk  = i - row * KdimPad;
    float v = (kk < Kdim) ? w[(size_t)row * Kdim + kk] : 0.0f;
    wbf[i] = (__bf16)v;
}

__global__ __launch_bounds__(256) void switched_conv_wmma(
    const float*  __restrict__ x,     // [B, IC, Hin, Win]
    const __bf16* __restrict__ wbf,   // [BR*OC, KdimPad] bf16, zero-padded
    const float*  __restrict__ bias,  // [OC]
    const float*  __restrict__ cw,    // [BR, IC]
    const float*  __restrict__ cb,    // [BR]
    const float*  __restrict__ resid, // [B, OC, Ho, Wo] or nullptr
    float* __restrict__ out,          // [B, OC, Ho, Wo]
    int IC, int OC, int K, int stride, int pad,
    int Hin, int Win, int Ho, int Wo,
    int Kdim, int KdimPad, int in_relu, int out_relu)
{
    __shared__ __bf16 As[2][128][LSTR];  // double-buffered weight tile
    __shared__ __bf16 Bst[2][BN][LSTR];  // double-buffered im2col tile [pixel][k]
    __shared__ float  Cs[128][BN];       // expert partial sums (64 KB)
    __shared__ float  Sel[BR][BN];       // routing weights
    __shared__ int    kd_cio[KD_MAX];    // kk -> ci*Hin*Win
    __shared__ int    kd_yx[KD_MAX];     // kk -> (ky<<16)|kx
    __shared__ int    px_base[BN], px_iy[BN], px_ix[BN], px_out[BN];

    const int tid  = threadIdx.x;
    const int wave = tid >> 5;           // expert id
    const int lane = tid & 31;
    const int half = lane >> 4;
    const int l16  = lane & 15;

    const int HoWo   = Ho * Wo;
    const int Ksteps = KdimPad / BK;
    const int n_base = blockIdx.x * BN;
    const int c_base = blockIdx.y * 16;

    // ---- prologue: pixel + k decode tables (hoists all div/mod out of hot loop)
    if (tid < BN) {
        int n  = n_base + tid;
        int b  = n / HoWo;
        int pr = n - b * HoWo;
        int oy = pr / Wo;
        int ox = pr - oy * Wo;
        px_base[tid] = b * IC * Hin * Win;
        px_iy[tid]   = oy * stride - pad;
        px_ix[tid]   = ox * stride - pad;
        px_out[tid]  = b * OC * HoWo + oy * Wo + ox;
    }
    for (int kk = tid; kk < Kdim; kk += 256) {
        int ci  = kk / (K * K);
        int rem = kk - ci * K * K;
        int ky  = rem / K;
        int kx  = rem - ky * K;
        kd_cio[kk] = ci * Hin * Win;
        kd_yx[kk]  = (ky << 16) | kx;
    }
    __syncthreads();

    // ---- tile staging (A via async-to-LDS, B via gather + packed b32 stores)
    auto stage = [&](int buf, int ks) {
        const int kk0 = ks * BK;
        #pragma unroll
        for (int i = 0; i < 2; ++i) {            // A: 512 chunks of 8 bf16
            int idx  = tid + i * 256;
            int row  = idx >> 2;
            int seg  = (idx & 3) << 3;
            int e    = row >> 4;
            int grow = e * OC + c_base + (row & 15);
            const __bf16* gp = wbf + (size_t)grow * KdimPad + kk0 + seg;
            unsigned ldsoff = (unsigned)(size_t)&As[buf][row][seg];
            asm volatile("global_load_async_to_lds_b128 %0, %1, off"
                         :: "v"(ldsoff), "v"(gp) : "memory");
        }
        #pragma unroll
        for (int i = 0; i < (BK * BN) / 512; ++i) {  // B: 8 bf16-pairs / thread
            int idx = tid + i * 256;
            int nc  = idx >> 4;
            int kr  = (idx & 15) << 1;
            int pb  = px_base[nc], iy0 = px_iy[nc], ix0 = px_ix[nc];
            float v0 = 0.0f, v1 = 0.0f;
            #pragma unroll
            for (int j = 0; j < 2; ++j) {
                int kk = kk0 + kr + j;
                float vv = 0.0f;
                if (kk < Kdim) {
                    int yx = kd_yx[kk];
                    int iy = iy0 + (yx >> 16);
                    int ix = ix0 + (yx & 0xffff);
                    if ((unsigned)iy < (unsigned)Hin && (unsigned)ix < (unsigned)Win) {
                        vv = x[(size_t)(pb + kd_cio[kk] + iy * Win + ix)];
                        if (in_relu) vv = fmaxf(vv, 0.0f);
                    }
                }
                if (j == 0) v0 = vv; else v1 = vv;
            }
            union { __bf16 h[2]; unsigned u; } pk;
            pk.h[0] = (__bf16)v0; pk.h[1] = (__bf16)v1;
            *(unsigned*)&Bst[buf][nc][kr] = pk.u;
        }
    };

    v8f acc[8] = {v8f{}, v8f{}, v8f{}, v8f{}, v8f{}, v8f{}, v8f{}, v8f{}};

    stage(0, 0);
    for (int ks = 0; ks < Ksteps; ++ks) {
        const int cur = ks & 1;
        asm volatile("s_wait_asynccnt 0x0" ::: "memory");  // tile `cur` A-data landed
        __syncthreads();                                   // ...and visible to all waves
        if (ks + 1 < Ksteps)
            stage(cur ^ 1, ks + 1);                        // overlap next tile with math

        // ---- fragments via b128 LDS loads (ISA reg layout -> contiguous 16B chunks)
        v8bf alo = *(const v8bf*)&As[cur][wave * 16 + l16][half * 8];
        v8bf ahi = *(const v8bf*)&As[cur][wave * 16 + l16][16 + half * 8];
        v16bf afrag = __builtin_shufflevector(alo, ahi,
            0,1,2,3,4,5,6,7,8,9,10,11,12,13,14,15);
        #pragma unroll
        for (int t = 0; t < 8; ++t) {
            v8bf blo = *(const v8bf*)&Bst[cur][t * 16 + l16][half * 8];
            v8bf bhi = *(const v8bf*)&Bst[cur][t * 16 + l16][16 + half * 8];
            v16bf bfrag = __builtin_shufflevector(blo, bhi,
                0,1,2,3,4,5,6,7,8,9,10,11,12,13,14,15);
            acc[t] = __builtin_amdgcn_wmma_f32_16x16x32_bf16(
                false, afrag, false, bfrag, (short)0, acc[t], false, false);
        }
    }
    __syncthreads();

    // ---- spill expert partials (C layout: M = e + 8*half, N = l16)
    #pragma unroll
    for (int t = 0; t < 8; ++t)
        #pragma unroll
        for (int e = 0; e < 8; ++e)
            Cs[wave * 16 + half * 8 + e][t * 16 + l16] = acc[t][e];
    __syncthreads();

    // ---- coupler: per-pixel routing logits + softmax (threads 0..127)
    if (tid < BN) {
        int pb = px_base[tid];
        int cy = px_iy[tid] + pad, cx = px_ix[tid] + pad;  // 1x1 VALID stride conv
        int xo = pb + cy * Win + cx;
        float logit[BR];
        #pragma unroll
        for (int e = 0; e < BR; ++e) logit[e] = cb[e];
        for (int ci = 0; ci < IC; ++ci) {
            float xv = x[(size_t)xo + ci * Hin * Win];
            if (in_relu) xv = fmaxf(xv, 0.0f);
            #pragma unroll
            for (int e = 0; e < BR; ++e) logit[e] += cw[e * IC + ci] * xv;
        }
        float mx = logit[0];
        #pragma unroll
        for (int e = 1; e < BR; ++e) mx = fmaxf(mx, logit[e]);
        float s = 0.0f;
        #pragma unroll
        for (int e = 0; e < BR; ++e) { logit[e] = __expf(logit[e] - mx); s += logit[e]; }
        float inv = 1.0f / s;
        #pragma unroll
        for (int e = 0; e < BR; ++e) Sel[e][tid] = logit[e] * inv;
    }
    __syncthreads();

    // ---- route + bias + residual + relu + store: 16*128 outputs, 8 / thread
    #pragma unroll
    for (int i = 0; i < (16 * BN) / 256; ++i) {
        int idx = tid + i * 256;
        int ci  = idx >> 7, nc = idx & (BN - 1);
        int c   = c_base + ci;
        float v = bias[c];
        #pragma unroll
        for (int e = 0; e < BR; ++e)
            v += Cs[e * 16 + ci][nc] * Sel[e][nc];
        size_t oidx = (size_t)px_out[nc] + (size_t)c * HoWo;
        if (resid) v += resid[oidx];
        if (out_relu) v = fmaxf(v, 0.0f);
        out[oidx] = v;
    }
}

// ---------------- host side ----------------

static inline int kpad32(int k) { return (k + 31) & ~31; }

static inline void run_cvt(const void* w, __bf16* wbf, int Mrows, int Kdim,
                           hipStream_t stream) {
    int KP = kpad32(Kdim);
    int total = Mrows * KP;
    convert_weights<<<(total + 255) / 256, 256, 0, stream>>>(
        (const float*)w, wbf, Mrows, Kdim, KP);
}

static inline void run_sc(const float* x, const __bf16* wbf, const void* const* p4,
                          const float* resid, float* out,
                          int Bn, int IC, int OC, int K, int stride, int pad,
                          int Hin, int Win, int Ho, int Wo,
                          int in_relu, int out_relu, hipStream_t stream)
{
    const float* b  = (const float*)p4[1];
    const float* cw = (const float*)p4[2];
    const float* cb = (const float*)p4[3];
    int Kdim = IC * K * K;
    dim3 grid((Bn * Ho * Wo) / BN, OC / 16);
    switched_conv_wmma<<<grid, 256, 0, stream>>>(
        x, wbf, b, cw, cb, resid, out,
        IC, OC, K, stride, pad, Hin, Win, Ho, Wo,
        Kdim, kpad32(Kdim), in_relu, out_relu);
}

extern "C" void kernel_launch(void* const* d_in, const int* in_sizes, int n_in,
                              void* d_out, int out_size, void* d_ws, size_t ws_size,
                              hipStream_t stream) {
    (void)in_sizes; (void)n_in; (void)out_size; (void)ws_size;
    const int Bn = 4;
    const float* x = (const float*)d_in[0];
    const void* const* P[7] = { &d_in[1], &d_in[5], &d_in[9], &d_in[13],
                                &d_in[17], &d_in[21], &d_in[25] };

    // ---- workspace carve: activation f32 buffers, then bf16 weight buffers
    float* ws = (float*)d_ws;
    float* a1 = ws;                       // [4,64,128,128]  = 4,194,304 f
    float* a2 = a1 + 4194304;             // [4,128,64,64]   = 2,097,152 f
    float* h0 = a2 + 2097152;             // [4,128,64,64]
    float* rr = h0 + 2097152;             // [4,32,64,64]    = 524,288 f
    float* h1 = a1;                       // reuse a1 after L1 consumed it

    __bf16* wb = (__bf16*)(rr + 524288);
    const int MR[7] = {512, 1024, 1024, 256, 1024, 256, 1024};
    const int KD[7] = {75, 1600, 1152, 1152, 32, 1152, 32};
    __bf16* WB[7];
    {
        __bf16* p = wb;
        for (int i = 0; i < 7; ++i) { WB[i] = p; p += (size_t)MR[i] * kpad32(KD[i]); }
    }
    for (int i = 0; i < 7; ++i)
        run_cvt(P[i][0], WB[i], MR[i], KD[i], stream);

    // L0: 3 -> 64, k5 s2 p2, relu
    run_sc(x,  WB[0], P[0], nullptr, a1, Bn,   3,  64, 5, 2, 2, 256, 256, 128, 128, 0, 1, stream);
    // L1: 64 -> 128, k5 s2 p2, relu
    run_sc(a1, WB[1], P[1], nullptr, a2, Bn,  64, 128, 5, 2, 2, 128, 128,  64,  64, 0, 1, stream);
    // L2: 128 -> 128, k3 s1 p1
    run_sc(a2, WB[2], P[2], nullptr, h0, Bn, 128, 128, 3, 1, 1,  64,  64,  64,  64, 0, 0, stream);
    // Res block 0
    run_sc(h0, WB[3], P[3], nullptr, rr, Bn, 128,  32, 3, 1, 1,  64,  64,  64,  64, 1, 1, stream);
    run_sc(rr, WB[4], P[4], h0,      h1, Bn,  32, 128, 1, 1, 0,  64,  64,  64,  64, 0, 0, stream);
    // Res block 1 (+ final relu)
    run_sc(h1, WB[5], P[5], nullptr, rr, Bn, 128,  32, 3, 1, 1,  64,  64,  64,  64, 1, 1, stream);
    run_sc(rr, WB[6], P[6], h1, (float*)d_out, Bn, 32, 128, 1, 1, 0, 64, 64, 64, 64, 0, 1, stream);
}
